// TransformerBlock_26396869001643
// MI455X (gfx1250) — compile-verified, matrix operand order checked by reading, not run on hardware
//
#include <hip/hip_runtime.h>
#include <hip/hip_bf16.h>
#include <math.h>

// ---------------- problem constants ----------------
#define S_LEN 2048
#define DMODEL 1024
#define NH 16
#define HD 64
#define BATCH 4
#define DFF 4096
#define MROWS (BATCH * S_LEN)   // 8192
#define EPS 1e-5f

// ---------------- types ----------------
typedef __attribute__((ext_vector_type(16))) __bf16 v16bf;
typedef __attribute__((ext_vector_type(8)))  __bf16 v8bf;
typedef __attribute__((ext_vector_type(8)))  float  v8f;

static __device__ __forceinline__ __bf16 f2bf(float f) {
    union { float f; unsigned int u; } c; c.f = f;
    unsigned int u = c.u + 0x7FFFu + ((c.u >> 16) & 1u);
    union { unsigned short s; __bf16 b; } o; o.s = (unsigned short)(u >> 16);
    return o.b;
}

static __device__ __forceinline__ v8f zero8() {
    v8f z;
#pragma unroll
    for (int i = 0; i < 8; ++i) z[i] = 0.0f;
    return z;
}

static __device__ __forceinline__ v8f wmma_bf16(v16bf a, v16bf b, v8f c) {
    return __builtin_amdgcn_wmma_f32_16x16x32_bf16(
        false, a, false, b, (short)0, c, false, false);
}

// Load a 16x32 bf16 operand fragment from a row-major row pointer.
// rowptr points at (row, k).  off = (lane>>4)*8.
// elements 0..7  <- K = off .. off+7
// elements 8..15 <- K = 16+off .. 16+off+7   (CDNA5 16-bit operand layout)
static __device__ __forceinline__ v16bf load_frag(const __bf16* rowptr, int off) {
    v8bf lo = *(const v8bf*)(rowptr + off);
    v8bf hi = *(const v8bf*)(rowptr + off + 16);
    return __builtin_shufflevector(lo, hi,
        0, 1, 2, 3, 4, 5, 6, 7, 8, 9, 10, 11, 12, 13, 14, 15);
}

// ---------------- weight convert: f32 [K,N] -> bf16 [N,K] (transposed) -------
__global__ void convert_w_kernel(const float* __restrict__ w,
                                 __bf16* __restrict__ wt, int K, int N) {
    size_t idx = (size_t)blockIdx.x * 256 + threadIdx.x;
    if (idx >= (size_t)K * N) return;
    int kk = (int)(idx / N);
    int nn = (int)(idx % N);
    wt[(size_t)nn * K + kk] = f2bf(w[idx]);
}

// ---------------- layernorm: f32 row -> bf16 row -----------------------------
__global__ void ln_kernel(const float* __restrict__ x,
                          const float* __restrict__ g,
                          const float* __restrict__ s,
                          __bf16* __restrict__ out) {
    int row = blockIdx.x;
    const float* xr = x + (size_t)row * DMODEL;
    float sum = 0.f, sq = 0.f;
    for (int i = threadIdx.x; i < DMODEL; i += 256) {
        float v = xr[i]; sum += v; sq += v * v;
    }
#pragma unroll
    for (int m = 16; m >= 1; m >>= 1) {
        sum += __shfl_xor(sum, m, 32);
        sq  += __shfl_xor(sq,  m, 32);
    }
    __shared__ float ssum[8], ssq[8];
    int wid = threadIdx.x >> 5, lane = threadIdx.x & 31;
    if (lane == 0) { ssum[wid] = sum; ssq[wid] = sq; }
    __syncthreads();
    float ts = 0.f, tq = 0.f;
#pragma unroll
    for (int i = 0; i < 8; ++i) { ts += ssum[i]; tq += ssq[i]; }
    float mean = ts / (float)DMODEL;
    float var  = tq / (float)DMODEL - mean * mean;
    float rstd = rsqrtf(var + EPS);
    __bf16* outr = out + (size_t)row * DMODEL;
    for (int i = threadIdx.x; i < DMODEL; i += 256) {
        outr[i] = f2bf(g[i] * (xr[i] - mean) * rstd + s[i]);
    }
}

// ---------------- WMMA GEMM (software-pipelined K loop) -----------------------
// C[M,N] = A[M,K] (bf16, row-major) * Bt[N,K]^T (bf16, pre-transposed weights)
// mode 0: f32 out (+bias)(+residual)      mode 1: bf16 out (+bias)
// mode 2: bf16 out + bias + GELU          mode 3: bf16 out scattered as vt[b,h,hd,s]
// __launch_bounds__(256, 1): favor VGPR budget over occupancy so the
// double-buffered fragments + 8 accumulator tiles fit without spilling.
__global__ void __launch_bounds__(256, 1)
gemm_kernel(const __bf16* __restrict__ A,
            const __bf16* __restrict__ Bt,
            void* __restrict__ outp,
            const float* __restrict__ bias,
            const float* __restrict__ res,
            int M, int N, int K, int mode) {
    int lane = threadIdx.x & 31, wid = threadIdx.x >> 5;
    int wm = wid >> 1, wn = wid & 1;         // 4 x 2 wave grid in block
    int m0 = blockIdx.y * 128 + wm * 32;
    int n0 = blockIdx.x * 128 + wn * 64;
    int half = lane >> 4, ln = lane & 15, off = half * 8;

    v8f acc[2][4];
#pragma unroll
    for (int i = 0; i < 2; ++i)
#pragma unroll
        for (int j = 0; j < 4; ++j) acc[i][j] = zero8();

    const __bf16* arow0 = A + (size_t)(m0 + ln)      * K;
    const __bf16* arow1 = A + (size_t)(m0 + 16 + ln) * K;
    const __bf16* brow0 = Bt + (size_t)(n0 + ln)      * K;
    const __bf16* brow1 = Bt + (size_t)(n0 + 16 + ln) * K;
    const __bf16* brow2 = Bt + (size_t)(n0 + 32 + ln) * K;
    const __bf16* brow3 = Bt + (size_t)(n0 + 48 + ln) * K;

    // prologue: fragments for k = 0
    v16bf a0c = load_frag(arow0, off);
    v16bf a1c = load_frag(arow1, off);
    v16bf b0c = load_frag(brow0, off);
    v16bf b1c = load_frag(brow1, off);
    v16bf b2c = load_frag(brow2, off);
    v16bf b3c = load_frag(brow3, off);

    // pipelined main loop: issue k+32 loads before consuming k fragments
    for (int k = 32; k < K; k += 32) {
        v16bf a0n = load_frag(arow0 + k, off);
        v16bf a1n = load_frag(arow1 + k, off);
        v16bf b0n = load_frag(brow0 + k, off);
        v16bf b1n = load_frag(brow1 + k, off);
        v16bf b2n = load_frag(brow2 + k, off);
        v16bf b3n = load_frag(brow3 + k, off);

        acc[0][0] = wmma_bf16(a0c, b0c, acc[0][0]);
        acc[1][0] = wmma_bf16(a1c, b0c, acc[1][0]);
        acc[0][1] = wmma_bf16(a0c, b1c, acc[0][1]);
        acc[1][1] = wmma_bf16(a1c, b1c, acc[1][1]);
        acc[0][2] = wmma_bf16(a0c, b2c, acc[0][2]);
        acc[1][2] = wmma_bf16(a1c, b2c, acc[1][2]);
        acc[0][3] = wmma_bf16(a0c, b3c, acc[0][3]);
        acc[1][3] = wmma_bf16(a1c, b3c, acc[1][3]);

        a0c = a0n; a1c = a1n;
        b0c = b0n; b1c = b1n; b2c = b2n; b3c = b3n;
    }
    // epilogue of the pipeline: last k-step
    acc[0][0] = wmma_bf16(a0c, b0c, acc[0][0]);
    acc[1][0] = wmma_bf16(a1c, b0c, acc[1][0]);
    acc[0][1] = wmma_bf16(a0c, b1c, acc[0][1]);
    acc[1][1] = wmma_bf16(a1c, b1c, acc[1][1]);
    acc[0][2] = wmma_bf16(a0c, b2c, acc[0][2]);
    acc[1][2] = wmma_bf16(a1c, b2c, acc[1][2]);
    acc[0][3] = wmma_bf16(a0c, b3c, acc[0][3]);
    acc[1][3] = wmma_bf16(a1c, b3c, acc[1][3]);

    // store epilogue
#pragma unroll
    for (int i = 0; i < 2; ++i) {
#pragma unroll
        for (int j = 0; j < 4; ++j) {
#pragma unroll
            for (int r = 0; r < 8; ++r) {
                int m = m0 + 16 * i + r + 8 * half;
                int n = n0 + 16 * j + ln;
                float v = acc[i][j][r];
                if (bias) v += bias[n];
                if (mode == 2) {
                    float t = v + 0.044715f * v * v * v;
                    v = 0.5f * v * (1.0f + tanhf(0.7978845608028654f * t));
                }
                if (mode == 0) {
                    if (res) v += res[(size_t)m * N + n];
                    ((float*)outp)[(size_t)m * N + n] = v;
                } else if (mode == 3) {
                    int bb = m / S_LEN, ss = m % S_LEN;
                    int hh = n / HD,   hd = n % HD;
                    ((__bf16*)outp)[(((size_t)(bb * NH + hh)) * HD + hd) * S_LEN + ss] = f2bf(v);
                } else {
                    ((__bf16*)outp)[(size_t)m * N + n] = f2bf(v);
                }
            }
        }
    }
}

// ---------------- flash attention (causal) ------------------------------------
// One wave per 16-row q tile of one (b,h).  Scores + PV via WMMA; P is
// re-laid out C->A through an LDS tile.  vt is head-transposed [b,h,hd,s].
// V fragment loads are issued BEFORE the softmax VALU block so the exp/shuffle
// work hides their latency.
__global__ void __launch_bounds__(256, 1)
attn_kernel(const __bf16* __restrict__ q,
            const __bf16* __restrict__ k,
            const __bf16* __restrict__ vt,
            __bf16* __restrict__ ctx) {
    __shared__ __align__(16) __bf16 pbuf[8][16 * 32];
    int lane = threadIdx.x & 31, wid = threadIdx.x >> 5;
    int t  = blockIdx.x * 8 + wid;
    int qt = t & (S_LEN / 16 - 1);
    int bh = t >> 7;
    int h  = bh & (NH - 1);
    int b  = bh >> 4;
    int q0 = qt * 16;
    int half = lane >> 4, ln = lane & 15, off = half * 8;
    const float scale = 0.125f;  // 1/sqrt(64)

    const __bf16* qrow = q + ((size_t)(b * S_LEN + q0 + ln)) * DMODEL + h * HD;
    v16bf aq0 = load_frag(qrow, off);        // hd 0..31
    v16bf aq1 = load_frag(qrow + 32, off);   // hd 32..63

    float mrow[8], lrow[8];
    v8f o[4];
#pragma unroll
    for (int r = 0; r < 8; ++r) { mrow[r] = -1e30f; lrow[r] = 0.f; }
#pragma unroll
    for (int j = 0; j < 4; ++j) o[j] = zero8();

    __bf16* pb = &pbuf[wid][0];

    for (int kv0 = 0; kv0 < q0 + 16; kv0 += 32) {
        // ---- scores: S = Q Kt, two 16x16 tiles covering kv0..kv0+31
        int kr0 = kv0 + ln;       if (kr0 > S_LEN - 1) kr0 = S_LEN - 1;
        int kr1 = kv0 + 16 + ln;  if (kr1 > S_LEN - 1) kr1 = S_LEN - 1;
        const __bf16* kp0 = k + ((size_t)(b * S_LEN + kr0)) * DMODEL + h * HD;
        const __bf16* kp1 = k + ((size_t)(b * S_LEN + kr1)) * DMODEL + h * HD;
        v8f s0 = zero8(), s1 = zero8();
        {
            v16bf b00 = load_frag(kp0, off), b01 = load_frag(kp0 + 32, off);
            v16bf b10 = load_frag(kp1, off), b11 = load_frag(kp1 + 32, off);
            s0 = wmma_bf16(aq0, b00, s0); s0 = wmma_bf16(aq1, b01, s0);
            s1 = wmma_bf16(aq0, b10, s1); s1 = wmma_bf16(aq1, b11, s1);
        }
        // ---- prefetch V fragments (independent of softmax) while exp/shuffle run
        v16bf bv[4];
#pragma unroll
        for (int j = 0; j < 4; ++j) {
            const __bf16* vr = vt + ((size_t)(bh * HD + j * 16 + ln)) * S_LEN + kv0;
            bv[j] = load_frag(vr, off);
        }
        // ---- online softmax update (row stats replicated per 16-lane half)
#pragma unroll
        for (int r = 0; r < 8; ++r) {
            int row = q0 + r + 8 * half;
            float v0 = s0[r] * scale; if (kv0 + ln > row)       v0 = -1e30f;
            float v1 = s1[r] * scale; if (kv0 + 16 + ln > row)  v1 = -1e30f;
            float rm = fmaxf(v0, v1);
#pragma unroll
            for (int m = 8; m >= 1; m >>= 1) rm = fmaxf(rm, __shfl_xor(rm, m, 32));
            float mn    = fmaxf(mrow[r], rm);
            float alpha = __expf(mrow[r] - mn);
            float p0 = __expf(v0 - mn), p1 = __expf(v1 - mn);
            float ps = p0 + p1;
#pragma unroll
            for (int m = 8; m >= 1; m >>= 1) ps += __shfl_xor(ps, m, 32);
            lrow[r] = lrow[r] * alpha + ps;
            mrow[r] = mn;
#pragma unroll
            for (int j = 0; j < 4; ++j) o[j][r] *= alpha;
            // stage P into LDS (C-layout write)
            pb[(r + 8 * half) * 32 + ln]      = f2bf(p0);
            pb[(r + 8 * half) * 32 + 16 + ln] = f2bf(p1);
        }
        // ---- reload P as an A fragment (row = ln, K = 32 kv values)
        const __bf16* pr = pb + ln * 32;
        v8bf plo = *(const v8bf*)(pr + off);
        v8bf phi = *(const v8bf*)(pr + 16 + off);
        v16bf ap = __builtin_shufflevector(plo, phi,
            0, 1, 2, 3, 4, 5, 6, 7, 8, 9, 10, 11, 12, 13, 14, 15);
        // ---- O += P V  (vt rows = hd channels, contiguous along s)
#pragma unroll
        for (int j = 0; j < 4; ++j)
            o[j] = wmma_bf16(ap, bv[j], o[j]);
    }

    // ---- finalize: O /= l, write ctx bf16 [b,s,d]
#pragma unroll
    for (int r = 0; r < 8; ++r) {
        float inv = 1.0f / lrow[r];
        int row = q0 + r + 8 * half;
        size_t base = ((size_t)(b * S_LEN + row)) * DMODEL + h * HD;
#pragma unroll
        for (int j = 0; j < 4; ++j)
            ctx[base + j * 16 + ln] = f2bf(o[j][r] * inv);
    }
}

// ---------------- host orchestration -----------------------------------------
extern "C" void kernel_launch(void* const* d_in, const int* in_sizes, int n_in,
                              void* d_out, int out_size, void* d_ws, size_t ws_size,
                              hipStream_t stream) {
    (void)in_sizes; (void)n_in; (void)out_size; (void)ws_size;
    const float* x  = (const float*)d_in[0];
    const float* Wq = (const float*)d_in[1];
    const float* Wk = (const float*)d_in[2];
    const float* Wv = (const float*)d_in[3];
    const float* Wo = (const float*)d_in[4];
    const float* bo = (const float*)d_in[5];
    const float* W1 = (const float*)d_in[6];
    const float* b1 = (const float*)d_in[7];
    const float* W2 = (const float*)d_in[8];
    const float* b2 = (const float*)d_in[9];
    const float* g1 = (const float*)d_in[10];
    const float* s1 = (const float*)d_in[11];
    const float* g2 = (const float*)d_in[12];
    const float* s2 = (const float*)d_in[13];
    float* out = (float*)d_out;

    // workspace carving
    unsigned char* base = (unsigned char*)d_ws;
    size_t off = 0;
    auto alloc = [&](size_t bytes) -> unsigned char* {
        unsigned char* p = base + off;
        off += (bytes + 255) & ~(size_t)255;
        return p;
    };
    const size_t actsBF = (size_t)MROWS * DMODEL * sizeof(__bf16);   // 16 MB
    __bf16* xn  = (__bf16*)alloc(actsBF);
    __bf16* qb  = (__bf16*)alloc(actsBF);
    __bf16* kb  = (__bf16*)alloc(actsBF);
    __bf16* vt  = (__bf16*)alloc(actsBF + 4096);  // + pad for causal-tail reads
    __bf16* ctx = (__bf16*)alloc(actsBF);
    float*  hb  = (float*) alloc((size_t)MROWS * DMODEL * sizeof(float));
    __bf16* yb  = (__bf16*)alloc(actsBF);
    __bf16* f1  = (__bf16*)alloc((size_t)MROWS * DFF * sizeof(__bf16));
    __bf16* wqt = (__bf16*)alloc((size_t)DMODEL * DMODEL * sizeof(__bf16));
    __bf16* wkt = (__bf16*)alloc((size_t)DMODEL * DMODEL * sizeof(__bf16));
    __bf16* wvt = (__bf16*)alloc((size_t)DMODEL * DMODEL * sizeof(__bf16));
    __bf16* wot = (__bf16*)alloc((size_t)DMODEL * DMODEL * sizeof(__bf16));
    __bf16* w1t = (__bf16*)alloc((size_t)DMODEL * DFF    * sizeof(__bf16));
    __bf16* w2t = (__bf16*)alloc((size_t)DFF    * DMODEL * sizeof(__bf16));

    // 1) weight conversion / transpose (fp32 [K,N] -> bf16 [N,K])
    {
        int n1 = DMODEL * DMODEL;
        int n2 = DMODEL * DFF;
        convert_w_kernel<<<(n1 + 255) / 256, 256, 0, stream>>>(Wq, wqt, DMODEL, DMODEL);
        convert_w_kernel<<<(n1 + 255) / 256, 256, 0, stream>>>(Wk, wkt, DMODEL, DMODEL);
        convert_w_kernel<<<(n1 + 255) / 256, 256, 0, stream>>>(Wv, wvt, DMODEL, DMODEL);
        convert_w_kernel<<<(n1 + 255) / 256, 256, 0, stream>>>(Wo, wot, DMODEL, DMODEL);
        convert_w_kernel<<<(n2 + 255) / 256, 256, 0, stream>>>(W1, w1t, DMODEL, DFF);
        convert_w_kernel<<<(n2 + 255) / 256, 256, 0, stream>>>(W2, w2t, DFF, DMODEL);
    }

    // 2) LN1: xn = LN(x; g1,s1)  (bf16)
    ln_kernel<<<MROWS, 256, 0, stream>>>(x, g1, s1, xn);

    // 3) projections
    dim3 gProj(DMODEL / 128, MROWS / 128);   // (8, 64)
    gemm_kernel<<<gProj, 256, 0, stream>>>(xn, wqt, qb, nullptr, nullptr,
                                           MROWS, DMODEL, DMODEL, 1);
    gemm_kernel<<<gProj, 256, 0, stream>>>(xn, wkt, kb, nullptr, nullptr,
                                           MROWS, DMODEL, DMODEL, 1);
    gemm_kernel<<<gProj, 256, 0, stream>>>(xn, wvt, vt, nullptr, nullptr,
                                           MROWS, DMODEL, DMODEL, 3);  // head-transposed

    // 4) attention: 8192 q-tiles, 8 waves/block
    attn_kernel<<<(BATCH * NH * (S_LEN / 16)) / 8, 256, 0, stream>>>(qb, kb, vt, ctx);

    // 5) O projection + bias + residual(x) -> h (f32)
    gemm_kernel<<<gProj, 256, 0, stream>>>(ctx, wot, hb, bo, x,
                                           MROWS, DMODEL, DMODEL, 0);

    // 6) LN2 -> y (bf16)
    ln_kernel<<<MROWS, 256, 0, stream>>>(hb, g2, s2, yb);

    // 7) FFN GEMM1 + bias + GELU -> f1 (bf16)
    dim3 gF1(DFF / 128, MROWS / 128);        // (32, 64)
    gemm_kernel<<<gF1, 256, 0, stream>>>(yb, w1t, f1, b1, nullptr,
                                         MROWS, DFF, DMODEL, 2);

    // 8) FFN GEMM2 + bias + residual(h) -> out (f32)
    gemm_kernel<<<gProj, 256, 0, stream>>>(f1, w2t, out, b2, hb,
                                           MROWS, DMODEL, DFF, 0);
}
